// DSI_515396076121
// MI455X (gfx1250) — matrix operand order repaired and checked by reading, not compiled, not run on hardware
//
#include <hip/hip_runtime.h>
#include <hip/hip_bf16.h>

// ---------------------------------------------------------------------------
// MI455X (gfx1250) fused particle-MLP network.
//
// Roofline: 162 GFLOP vs 32.5 MB HBM traffic -> compute bound -> all matmuls
// on v_wmma_f32_16x16x32_f16 (f16 in, f32 accum). Weights (<2 MB) pre-swizzled
// once into fragment-ready 32x16 f16 tiles in d_ws (32B/lane contiguous loads,
// L2/L0 resident). Each wave privately owns a 16-sample tile -> no barriers.
//
// Round-2 changes:
//  * only one embedding-type accumulator (32 VGPRs) live at a time; flushed to
//    the LDS latent buffer per type  -> kills the scratch spills seen in r1.
//  * prenet ping-pongs inside buf1 (cols 0-127 / 128-255) so buf0 is free to
//    hold the latent for the whole prenet phase.
//  * fast GELU: x - x*rcp(exp2(2.30221*u)+1), u = x(1+0.044715x^2)
//    (identical tanh-gelu algebra, 5 VALU + v_exp_f32 + v_rcp_f32, inf-safe)
//    instead of the branchy OCML tanhf.
// ---------------------------------------------------------------------------

typedef __attribute__((ext_vector_type(16))) _Float16 v16h;
typedef __attribute__((ext_vector_type(8)))  _Float16 v8h;
typedef __attribute__((ext_vector_type(8)))  float    v8f;

#define WAVES        8
#define RS           264                 // LDS row stride in halves (256 + 8 pad)
#define WAVE_LDS_H   (2 * 16 * RS)       // halves per wave (two 16xRS buffers)

// ---- swizzled-weight workspace offsets (in halves), tile = 512 halves ----
#define PRE_STRIDE   28672               // prenet t: W1 @ t*28672, W2 +4096, W3 +20480
#define NET1_OFF     86016               // 202x256 -> 7x16 tiles
#define NET2_OFF     143360              // 256x256 -> 8x16 tiles
#define NET3_OFF     208896              // 256x256 -> 8x16 tiles
#define NET4_OFF     274432              // 256x1   -> 8x1  tiles

__device__ __forceinline__ float gelu_fast(float x) {
  // 0.5*x*(1+tanh(0.79788456*(x+0.044715*x^3)))  ==  x - x/(e+1),
  // e = exp2(2*0.79788456*log2e * u),  u = x*(1 + 0.044715*x^2)
  const float u = x * __builtin_fmaf(0.044715f, x * x, 1.0f);
  const float e = __builtin_amdgcn_exp2f(2.302214179f * u);
  const float r = __builtin_amdgcn_rcpf(e + 1.0f);
  return __builtin_fmaf(-x, r, x);       // x->+inf: e=inf,r=0 -> x ; x->-inf -> 0
}

__device__ __forceinline__ v8f wmma_f16(v16h a, v16h b, v8f c) {
  // (neg_a, A, neg_b, B, c_mod, C, reuse_a, reuse_b)
  return __builtin_amdgcn_wmma_f32_16x16x32_f16(false, a, false, b, (short)0, c,
                                                false, false);
}

// Load one 16x32 f16 A fragment from an LDS activation buffer.
// Layout (matches swizzled B): lanes 0-15 hold row M=lane, K = 0..15; lanes
// 16-31 hold K = 16..31. Two contiguous 16B chunks per lane.
__device__ __forceinline__ v16h load_a_frag(const _Float16* src, int kc, int m,
                                            int hi) {
  const v8h lo = *(const v8h*)(src + m * RS + kc * 32 + hi * 16);
  const v8h hh = *(const v8h*)(src + m * RS + kc * 32 + hi * 16 + 8);
  v16h a;
#pragma unroll
  for (int i = 0; i < 8; ++i) { a[i] = lo[i]; a[8 + i] = hh[i]; }
  return a;
}

// One dense layer: LDS acts (Kp = KC*32) x swizzled weights -> LDS acts.
// D tile layout: lane holds col n=lane&15, rows m = j + (lane>>4)*8.
// src/dst may point at a column offset within a row (they only add `col`).
template <int KC, int NT, bool ACT>
__device__ __forceinline__ void mlp_layer(const _Float16* __restrict__ src,
                                          const _Float16* __restrict__ wt,
                                          const float* __restrict__ bias,
                                          _Float16* __restrict__ dst, int lane) {
  const int m = lane & 15, hi = lane >> 4;
  v16h a[KC];
#pragma unroll
  for (int kc = 0; kc < KC; ++kc) a[kc] = load_a_frag(src, kc, m, hi);

  for (int nt = 0; nt < NT; ++nt) {
    v8f c = {};
#pragma unroll
    for (int kc = 0; kc < KC; ++kc) {
      v16h b = *(const v16h*)(wt + (size_t)(kc * NT + nt) * 512 + lane * 16);
      c = wmma_f16(a[kc], b, c);
    }
    const float bv = bias[nt * 16 + m];
#pragma unroll
    for (int j = 0; j < 8; ++j) {
      float v = c[j] + bv;
      if (ACT) v = gelu_fast(v);
      dst[(j + hi * 8) * RS + nt * 16 + m] = (_Float16)v;
    }
  }
}

struct Params {
  const float* x;
  const _Float16* w;          // swizzled f16 weight tiles in d_ws
  const float* pb1[3];        // prenet biases (f32, straight from inputs)
  const float* pb2[3];
  const float* pb3[3];
  const float* nb[4];         // net biases
  float* out;
  int nSamples;
};

__global__ void __launch_bounds__(WAVES * 32)
fused_particle_net(Params P) {
  extern __shared__ _Float16 smem[];
  const int lane = threadIdx.x & 31;
  const int wave = threadIdx.x >> 5;
  const int m = lane & 15, hi = lane >> 4;
  const int s0 = blockIdx.x * (WAVES * 16) + wave * 16;   // base sample of tile
  _Float16* buf0 = smem + wave * WAVE_LDS_H;              // latent / net ping
  _Float16* buf1 = buf0 + 16 * RS;                        // prenet ping-pong / net pong

  // ---------------- prenets: per particle 4 -> 128 -> 128 -> 64 -------------
  // TYPE_TOKENS = [0,0,1,2,2]: particles of a type are consecutive, so only
  // one type's 16x64 f32 accumulator (4 tiles = 32 VGPRs) is live at a time.
  int part = 0;
  for (int t = 0; t < 3; ++t) {
    const int nParts = (t == 1) ? 1 : 2;
    const _Float16* w1 = P.w + t * PRE_STRIDE;
    const _Float16* w2 = w1 + 4096;
    const _Float16* w3 = w1 + 20480;

    v8f acc[4];
#pragma unroll
    for (int n = 0; n < 4; ++n) acc[n] = (v8f){};

    for (int pp = 0; pp < nParts; ++pp, ++part) {
      // L1: K=4 (zero-padded to 32). Only lanes 0-15 carry data (K=0..3).
      v16h a1 = {};
      if (hi == 0) {
        int srow = s0 + m;
        if (srow >= P.nSamples) srow = P.nSamples - 1;
        const float* xr = P.x + (size_t)srow * 30 + 10 + part * 4;
#pragma unroll
        for (int i = 0; i < 4; ++i) a1[i] = (_Float16)xr[i];
      }
      for (int nt = 0; nt < 8; ++nt) {
        v16h b = *(const v16h*)(w1 + nt * 512 + lane * 16);
        v8f c = {};
        c = wmma_f16(a1, b, c);
        const float bv = P.pb1[t][nt * 16 + m];
#pragma unroll
        for (int j = 0; j < 8; ++j)
          buf1[(j + hi * 8) * RS + nt * 16 + m] =
              (_Float16)gelu_fast(c[j] + bv);
      }

      // L2: 128 -> 128 (gelu), buf1 cols[0,128) -> buf1 cols[128,256)
      mlp_layer<4, 8, true>(buf1, w2, P.pb2[t], buf1 + 128, lane);

      // L3: 128 -> 64 (linear) from buf1 cols[128,256), accumulate via C chain
      {
        v16h a3[4];
#pragma unroll
        for (int kc = 0; kc < 4; ++kc) a3[kc] = load_a_frag(buf1 + 128, kc, m, hi);
#pragma unroll
        for (int nt = 0; nt < 4; ++nt) {
          v8f c = acc[nt];
#pragma unroll
          for (int kc = 0; kc < 4; ++kc) {
            v16h b = *(const v16h*)(w3 + (size_t)(kc * 4 + nt) * 512 + lane * 16);
            c = wmma_f16(a3[kc], b, c);
          }
          const float bv = P.pb3[t][nt * 16 + m];  // bias per particle instance
#pragma unroll
          for (int j = 0; j < 8; ++j) c[j] += bv;
          acc[nt] = c;
        }
      }
    }

    // flush this type's embedding sum into the latent (buf0 cols t*64..)
#pragma unroll
    for (int nt = 0; nt < 4; ++nt)
#pragma unroll
      for (int j = 0; j < 8; ++j)
        buf0[(j + hi * 8) * RS + t * 64 + nt * 16 + m] = (_Float16)acc[nt][j];
  }

  // ------------- latent tail: nonfv (cols 192..201) + zero pad to 224 -------
  if (hi == 0) {                     // lanes 0-15 cover all 16 rows
    int srow = s0 + m;
    if (srow >= P.nSamples) srow = P.nSamples - 1;
    const float* xr = P.x + (size_t)srow * 30;
#pragma unroll
    for (int i = 0; i < 10; ++i) buf0[m * RS + 192 + i] = (_Float16)xr[i];
#pragma unroll
    for (int i = 10; i < 32; ++i) buf0[m * RS + 192 + i] = (_Float16)0.0f;
  }

  // ---------------- trunk: 202 -> 256 -> 256 -> 256 -> 1 --------------------
  const _Float16* n1 = P.w + NET1_OFF;
  const _Float16* n2 = P.w + NET2_OFF;
  const _Float16* n3 = P.w + NET3_OFF;
  const _Float16* n4 = P.w + NET4_OFF;
  mlp_layer<7, 16, true>(buf0, n1, P.nb[0], buf1, lane);
  mlp_layer<8, 16, true>(buf1, n2, P.nb[1], buf0, lane);
  mlp_layer<8, 16, true>(buf0, n3, P.nb[2], buf1, lane);

  // final 256 -> 1: N padded to 16 (cols 1..15 are zero tiles)
  {
    v16h a4[8];
#pragma unroll
    for (int kc = 0; kc < 8; ++kc) a4[kc] = load_a_frag(buf1, kc, m, hi);
    v8f c = {};
#pragma unroll
    for (int kc = 0; kc < 8; ++kc) {
      v16h b = *(const v16h*)(n4 + (size_t)kc * 512 + lane * 16);
      c = wmma_f16(a4[kc], b, c);
    }
    if (m == 0) {                    // lanes 0 / 16 hold column n==0
      const float b4 = P.nb[3][0];
#pragma unroll
      for (int j = 0; j < 8; ++j) {
        const int s = s0 + j + hi * 8;
        if (s < P.nSamples) P.out[s] = c[j] + b4;
      }
    }
  }
}

// ---------------------------------------------------------------------------
// Weight swizzle: f32 KxN row-major -> f16 fragment tiles.
// Tile (kc, nt) holds 32 lanes x 16 halves; element (l, i):
//   k = kc*32 + (l>>4)*16 + i , n = nt*16 + (l&15)   (zero-padded OOB)
// so a wave later reads its whole B fragment as 32 contiguous bytes per lane.
// ---------------------------------------------------------------------------
__global__ void swizzle_weights(const float* __restrict__ W,
                                _Float16* __restrict__ dst, int K, int N,
                                int nNt, int total) {
  int idx = blockIdx.x * 256 + threadIdx.x;
  if (idx >= total) return;
  int tile = idx >> 9, e = idx & 511;
  int l = e >> 4, i = e & 15;
  int kc = tile / nNt, nt = tile - kc * nNt;
  int k = kc * 32 + (l >> 4) * 16 + i;
  int n = nt * 16 + (l & 15);
  float v = (k < K && n < N) ? W[(size_t)k * N + n] : 0.0f;
  dst[idx] = (_Float16)v;
}

// ---------------------------------------------------------------------------
// Input layout (setup_inputs order, dicts flatten Ws then bs):
//  0: x | 1+6t..6+6t: prenet t {W1,W2,W3,b1,b2,b3} | 19..22: net W1..W4 |
//  23..26: net b1..b4 | 27: type_token (constant, hardcoded)
// ---------------------------------------------------------------------------
extern "C" void kernel_launch(void* const* d_in, const int* in_sizes, int n_in,
                              void* d_out, int out_size, void* d_ws,
                              size_t ws_size, hipStream_t stream) {
  (void)in_sizes; (void)n_in; (void)ws_size;
  _Float16* ws = (_Float16*)d_ws;

  auto sw = [&](int inIdx, size_t dstOff, int K, int N) {
    int nKc = (K + 31) / 32, nNt = (N + 15) / 16;
    int total = nKc * nNt * 512;
    swizzle_weights<<<(total + 255) / 256, 256, 0, stream>>>(
        (const float*)d_in[inIdx], ws + dstOff, K, N, nNt, total);
  };

  for (int t = 0; t < 3; ++t) {
    const int b = 1 + t * 6;
    sw(b + 0, (size_t)t * PRE_STRIDE + 0,     4,   128);
    sw(b + 1, (size_t)t * PRE_STRIDE + 4096,  128, 128);
    sw(b + 2, (size_t)t * PRE_STRIDE + 20480, 128, 64);
  }
  sw(19, NET1_OFF, 202, 256);
  sw(20, NET2_OFF, 256, 256);
  sw(21, NET3_OFF, 256, 256);
  sw(22, NET4_OFF, 256, 1);

  Params P;
  P.x = (const float*)d_in[0];
  P.w = ws;
  for (int t = 0; t < 3; ++t) {
    P.pb1[t] = (const float*)d_in[1 + t * 6 + 3];
    P.pb2[t] = (const float*)d_in[1 + t * 6 + 4];
    P.pb3[t] = (const float*)d_in[1 + t * 6 + 5];
  }
  for (int i = 0; i < 4; ++i) P.nb[i] = (const float*)d_in[23 + i];
  P.out = (float*)d_out;
  P.nSamples = out_size;

  const int samplesPerBlock = WAVES * 16;                 // 128
  const int blocks = (out_size + samplesPerBlock - 1) / samplesPerBlock;
  const size_t shmem = (size_t)WAVES * WAVE_LDS_H * sizeof(_Float16); // 132 KB
  fused_particle_net<<<blocks, WAVES * 32, shmem, stream>>>(P);
}
